// myHeteroGATConvSample_4123168604767
// MI455X (gfx1250) — matrix operand order repaired and checked by reading, not compiled, not run on hardware
//
#include <hip/hip_runtime.h>
#include <hip/hip_bf16.h>
#include <math.h>

#define N_NODES 50000
#define N_EDGES 800000
#define T_TYPES 3
#define FIN     256
#define H_HEADS 4
#define D_DIM   64
#define HD      256   // H_HEADS * D_DIM
#define FE_DIM  64
#define NEG_SLOPE 0.2f

typedef __attribute__((ext_vector_type(16))) __bf16 v16bf;
typedef __attribute__((ext_vector_type(8)))  float  v8f;

// ---------------------------------------------------------------------------
// float atomic max via signed/unsigned int atomics (valid with -inf init)
// ---------------------------------------------------------------------------
__device__ __forceinline__ void atomicMaxF(float* addr, float val) {
    if (val >= 0.0f) atomicMax((int*)addr, __float_as_int(val));
    else             atomicMin((unsigned int*)addr, __float_as_uint(val));
}

// ---------------------------------------------------------------------------
// zero the aggregation output region (atomics accumulate into it)
// ---------------------------------------------------------------------------
__global__ void zero4_kernel(float4* __restrict__ p, int n4) {
    int i = blockIdx.x * 256 + threadIdx.x;
    if (i < n4) p[i] = make_float4(0.f, 0.f, 0.f, 0.f);
}

// ---------------------------------------------------------------------------
// ee[t][h] = sum_e ( (edge_emb[t] @ W_e)[h*FE+e] * attn_e[h][e] ), tiny
// ---------------------------------------------------------------------------
__global__ void ee_kernel(const float* __restrict__ edge_emb,
                          const float* __restrict__ W_e,
                          const float* __restrict__ attn_e,
                          float* __restrict__ ee) {
    int tid = threadIdx.x;
    if (tid >= T_TYPES * H_HEADS) return;
    int t = tid >> 2, h = tid & 3;
    float s = 0.0f;
    for (int e = 0; e < FE_DIM; ++e) {
        float ef = 0.0f;
        for (int k = 0; k < FE_DIM; ++k)
            ef += edge_emb[t * FE_DIM + k] * W_e[k * (H_HEADS * FE_DIM) + h * FE_DIM + e];
        s += ef * attn_e[h * FE_DIM + e];
    }
    ee[t * H_HEADS + h] = s;
}

// ---------------------------------------------------------------------------
// feat = x @ W via v_wmma_f32_16x16x32_bf16.
// Grid: 3125 blocks x 128 threads (4 waves). Wave w = head h; block = 16 rows.
// Each wave: 16x64 output tile = 4 WMMA col-subtiles, 8 K-steps.
// A frag (16x32 bf16): lane&15 = M row; c0 = (lane>>4)*8; elem j -> K=c0+j,
//                      elem 8+j -> K=c0+16+j.
// B frag (32x16 bf16): lane&15 = N column; elem j -> K=c0+(j&7)+16*(j>>3).
// C/D   (16x16 f32):   lane&15 = N; vgpr r -> M = r + 8*(lane>>4).
// ---------------------------------------------------------------------------
__global__ __launch_bounds__(128) void gemm_feat_kernel(
        const float* __restrict__ x, const float* __restrict__ W,
        float* __restrict__ feat) {
    const int head = threadIdx.x >> 5;          // wave id = head
    const int lane = threadIdx.x & 31;
    const int m0   = blockIdx.x * 16;
    const int n0   = head * 64;
    const int mrow = lane & 15;
    const int half = lane >> 4;
    const int c0   = half * 8;

    v8f acc[4] = {};

    const float* xrow = x + (size_t)(m0 + mrow) * FIN;

    for (int k0 = 0; k0 < FIN; k0 += 32) {
        // ---- A fragment (fp32 -> bf16 in registers) ----
        v16bf a;
#pragma unroll
        for (int j = 0; j < 8; ++j) {
            a[j]     = (__bf16)xrow[k0 + c0 + j];
            a[8 + j] = (__bf16)xrow[k0 + 16 + c0 + j];
        }
        // ---- B fragments: lane holds W col (n0+t*16+mrow), K in table order ----
        const float* wk = W + (size_t)k0 * HD + n0 + mrow;
#pragma unroll
        for (int t = 0; t < 4; ++t) {
            v16bf b;
#pragma unroll
            for (int j = 0; j < 16; ++j) b[j] = (__bf16)wk[(size_t)(c0 + (j & 7) + ((j >> 3) << 4)) * HD + t * 16];
            acc[t] = __builtin_amdgcn_wmma_f32_16x16x32_bf16(
                         false, a, false, b, (short)0, acc[t], false, false);
        }
    }
#pragma unroll
    for (int t = 0; t < 4; ++t) {
#pragma unroll
        for (int r = 0; r < 8; ++r) {
            feat[(size_t)(m0 + r + 8 * half) * HD + n0 + t * 16 + mrow] = acc[t][r];
        }
    }
}

// ---------------------------------------------------------------------------
// Per-node: el[n][h], er[n][h] reductions; init seg_max=-inf, seg_sum=0.
// 1 block per node, 256 threads (one per h*64+d), LDS tree reduce per 64-group.
// ---------------------------------------------------------------------------
__global__ __launch_bounds__(256) void node_elr_kernel(
        const float* __restrict__ feat,
        const float* __restrict__ attn_l, const float* __restrict__ attn_r,
        float* __restrict__ el, float* __restrict__ er,
        float* __restrict__ seg_max, float* __restrict__ seg_sum) {
    __shared__ float sl[256];
    __shared__ float sr[256];
    const int j = threadIdx.x;
    const int n = blockIdx.x;
    const float f = feat[(size_t)n * HD + j];
    sl[j] = f * attn_l[j];
    sr[j] = f * attn_r[j];
    __syncthreads();
#pragma unroll
    for (int s = 32; s > 0; s >>= 1) {
        if ((j & 63) < s) { sl[j] += sl[j + s]; sr[j] += sr[j + s]; }
        __syncthreads();
    }
    if ((j & 63) == 0) {
        const int h = j >> 6;
        el[n * H_HEADS + h] = sl[j];
        er[n * H_HEADS + h] = sr[j];
        seg_max[n * H_HEADS + h] = -__builtin_huge_valf();
        seg_sum[n * H_HEADS + h] = 0.0f;
    }
}

// ---------------------------------------------------------------------------
// Pass 1 over edges: logits + atomic segment max into seg_max[dst]
// ---------------------------------------------------------------------------
__global__ void edge_logits_kernel(
        const int* __restrict__ esrc, const int* __restrict__ edst,
        const int* __restrict__ etyp,
        const float* __restrict__ el, const float* __restrict__ er,
        const float* __restrict__ ee,
        float* __restrict__ logits, float* __restrict__ seg_max) {
    const int e = blockIdx.x * 256 + threadIdx.x;
    if (e >= N_EDGES) return;
    const int s = esrc[e], d = edst[e], t = etyp[e];
    const float4 l4 = *(const float4*)(el + (size_t)s * 4);
    const float4 r4 = *(const float4*)(er + (size_t)d * 4);
    const float4 e4 = *(const float4*)(ee + (size_t)t * 4);
    float v0 = l4.x + r4.x + e4.x; v0 = v0 > 0.0f ? v0 : NEG_SLOPE * v0;
    float v1 = l4.y + r4.y + e4.y; v1 = v1 > 0.0f ? v1 : NEG_SLOPE * v1;
    float v2 = l4.z + r4.z + e4.z; v2 = v2 > 0.0f ? v2 : NEG_SLOPE * v2;
    float v3 = l4.w + r4.w + e4.w; v3 = v3 > 0.0f ? v3 : NEG_SLOPE * v3;
    *(float4*)(logits + (size_t)e * 4) = make_float4(v0, v1, v2, v3);
    atomicMaxF(&seg_max[d * 4 + 0], v0);
    atomicMaxF(&seg_max[d * 4 + 1], v1);
    atomicMaxF(&seg_max[d * 4 + 2], v2);
    atomicMaxF(&seg_max[d * 4 + 3], v3);
}

// ---------------------------------------------------------------------------
// Pass 2 over edges: expl = exp(logit - seg_max[dst]); atomic-add seg_sum[dst]
// (overwrites logits buffer with expl)
// ---------------------------------------------------------------------------
__global__ void edge_exp_kernel(
        const int* __restrict__ edst,
        float* __restrict__ logits, const float* __restrict__ seg_max,
        float* __restrict__ seg_sum) {
    const int e = blockIdx.x * 256 + threadIdx.x;
    if (e >= N_EDGES) return;
    const int d = edst[e];
    const float4 lg = *(const float4*)(logits + (size_t)e * 4);
    const float4 mx = *(const float4*)(seg_max + (size_t)d * 4);
    float e0 = expf(lg.x - mx.x);
    float e1 = expf(lg.y - mx.y);
    float e2 = expf(lg.z - mx.z);
    float e3 = expf(lg.w - mx.w);
    *(float4*)(logits + (size_t)e * 4) = make_float4(e0, e1, e2, e3);
    atomicAdd(&seg_sum[d * 4 + 0], e0);
    atomicAdd(&seg_sum[d * 4 + 1], e1);
    atomicAdd(&seg_sum[d * 4 + 2], e2);
    atomicAdd(&seg_sum[d * 4 + 3], e3);
}

// ---------------------------------------------------------------------------
// Pass 3: a = expl/seg_sum[dst] (written to out_a); weighted scatter of
// feat[src] into out[dst]. One wave per edge. Gather with float4
// (global_load_b128): iteration it covers 128 consecutive floats, lane owns
// 4 of them; head index h = it*2 + (lane>>4). Prefetch the source row while
// the softmax-normalization operands are in flight.
// ---------------------------------------------------------------------------
__global__ __launch_bounds__(256) void edge_scatter_kernel(
        const int* __restrict__ esrc, const int* __restrict__ edst,
        const float* __restrict__ feat, const float* __restrict__ expl,
        const float* __restrict__ seg_sum,
        float* __restrict__ out, float* __restrict__ out_a) {
    const int e    = blockIdx.x * 8 + (threadIdx.x >> 5);
    const int lane = threadIdx.x & 31;
    const int src  = esrc[e];
    const int dst  = edst[e];
    const float4* fsrc4 = (const float4*)(feat + (size_t)src * HD);
    __builtin_prefetch(&fsrc4[lane], 0, 1);            // global_prefetch of row
    const float4 ex = *(const float4*)(expl + (size_t)e * 4);
    const float4 ss = *(const float4*)(seg_sum + (size_t)dst * 4);
    const float a0 = ex.x / ss.x, a1 = ex.y / ss.y;
    const float a2 = ex.z / ss.z, a3 = ex.w / ss.w;
    if (lane < 4) {
        float av = (lane == 0) ? a0 : (lane == 1) ? a1 : (lane == 2) ? a2 : a3;
        out_a[(size_t)e * 4 + lane] = av;
    }
    float* odst = out + (size_t)dst * HD;
    const float alo = (lane < 16) ? a0 : a1;           // h for it=0
    const float ahi = (lane < 16) ? a2 : a3;           // h for it=1
#pragma unroll
    for (int it = 0; it < 2; ++it) {
        const int   i = it * 128 + lane * 4;           // 4 floats, same head
        const float4 f = fsrc4[it * 32 + lane];
        const float  a = (it == 0) ? alo : ahi;
        atomicAdd(odst + i + 0, f.x * a);
        atomicAdd(odst + i + 1, f.y * a);
        atomicAdd(odst + i + 2, f.z * a);
        atomicAdd(odst + i + 3, f.w * a);
    }
}

// ---------------------------------------------------------------------------
extern "C" void kernel_launch(void* const* d_in, const int* in_sizes, int n_in,
                              void* d_out, int out_size, void* d_ws, size_t ws_size,
                              hipStream_t stream) {
    (void)in_sizes; (void)n_in; (void)out_size; (void)ws_size;
    const float* x        = (const float*)d_in[0];
    const int*   edge_src = (const int*)  d_in[1];
    const int*   edge_dst = (const int*)  d_in[2];
    const int*   edge_typ = (const int*)  d_in[3];
    const float* W        = (const float*)d_in[4];
    const float* edge_emb = (const float*)d_in[5];
    const float* W_e      = (const float*)d_in[6];
    const float* attn_l   = (const float*)d_in[7];
    const float* attn_r   = (const float*)d_in[8];
    const float* attn_e   = (const float*)d_in[9];

    float* out   = (float*)d_out;                       // [N, H, D]
    float* out_a = out + (size_t)N_NODES * HD;          // [E, H]

    // workspace layout (floats)
    float* ws    = (float*)d_ws;
    float* feat  = ws;                                  // N*256
    float* el    = feat + (size_t)N_NODES * HD;         // N*4
    float* er    = el   + (size_t)N_NODES * 4;          // N*4
    float* smax  = er   + (size_t)N_NODES * 4;          // N*4
    float* ssum  = smax + (size_t)N_NODES * 4;          // N*4
    float* eebuf = ssum + (size_t)N_NODES * 4;          // 64 (12 used)
    float* lgbuf = eebuf + 64;                          // E*4 (logits -> expl)

    // 1) zero aggregation output (atomics accumulate; graph is replayed)
    zero4_kernel<<<(N_NODES * HD / 4 + 255) / 256, 256, 0, stream>>>(
        (float4*)out, N_NODES * HD / 4);
    // 2) tiny per-edge-type attention scalars
    ee_kernel<<<1, 16, 0, stream>>>(edge_emb, W_e, attn_e, eebuf);
    // 3) node projection GEMM via WMMA (bf16 in, f32 accumulate)
    gemm_feat_kernel<<<N_NODES / 16, 128, 0, stream>>>(x, W, feat);
    // 4) per-node el/er + segment init
    node_elr_kernel<<<N_NODES, 256, 0, stream>>>(feat, attn_l, attn_r,
                                                 el, er, smax, ssum);
    // 5) edge logits + segment max
    edge_logits_kernel<<<N_EDGES / 256, 256, 0, stream>>>(edge_src, edge_dst, edge_typ,
                                                          el, er, eebuf, lgbuf, smax);
    // 6) exp + segment sum
    edge_exp_kernel<<<N_EDGES / 256, 256, 0, stream>>>(edge_dst, lgbuf, smax, ssum);
    // 7) normalize + weighted scatter-sum (wave per edge)
    edge_scatter_kernel<<<N_EDGES / 8, 256, 0, stream>>>(edge_src, edge_dst, feat,
                                                         lgbuf, ssum, out, out_a);
}
